// DilatedAttention_3530463117931
// MI455X (gfx1250) — matrix operand order repaired and checked by reading, not compiled, not run on hardware
//
#include <hip/hip_runtime.h>

typedef __attribute__((ext_vector_type(16))) __bf16 v16bf;
typedef __attribute__((ext_vector_type(8)))  float  v8f;
typedef __attribute__((ext_vector_type(2)))  __bf16 v2bf;

#define DHEAD 1024
#define SCALE 0.03125f   /* 1/sqrt(1024) */

#define PB_STRIDE   520          /* in bf16 units */
#define PBUF_BYTES  133120
#define SMEM_BYTES  (PBUF_BYTES + 40960)

union V16 { v16bf v; unsigned u[8]; };

__device__ __forceinline__ unsigned short f2bf(float f) {
  unsigned u = __builtin_bit_cast(unsigned, f);
  u = (u + 0x7FFFu + ((u >> 16) & 1u)) >> 16;
  return (unsigned short)u;
}
__device__ __forceinline__ unsigned f2bf2(float lo, float hi) {
  unsigned ul = __builtin_bit_cast(unsigned, lo);
  unsigned uh = __builtin_bit_cast(unsigned, hi);
  ul = (ul + 0x7FFFu + ((ul >> 16) & 1u)) >> 16;
  uh = (uh + 0x7FFFu + ((uh >> 16) & 1u)) >> 16;
  return ul | (uh << 16);
}

// ---- CDNA5 async global->LDS copy (per-lane, ASYNCcnt-tracked) ----
typedef const __attribute__((address_space(3))) unsigned char* lds_cptr;
__device__ __forceinline__ unsigned lds_offset_of(const void* p) {
  return (unsigned)(unsigned long long)(lds_cptr)p;
}
__device__ __forceinline__ void async_ld_b64(unsigned lds_off, const void* g) {
  asm volatile("global_load_async_to_lds_b64 %0, %1, off"
               :: "v"(lds_off), "v"(g) : "memory");
}
__device__ __forceinline__ void wait_async0() {
#if __has_builtin(__builtin_amdgcn_s_wait_asynccnt)
  __builtin_amdgcn_s_wait_asynccnt(0);
#else
  asm volatile("s_wait_asynccnt 0" ::: "memory");
#endif
}

// scheduling fence: nothing may cross (pins the software pipeline)
#if __has_builtin(__builtin_amdgcn_sched_barrier)
#define SCHED_FENCE() __builtin_amdgcn_sched_barrier(0)
#else
#define SCHED_FENCE() do {} while (0)
#endif

// 16B-aligned LDS read of 16 packed bf16 (one WMMA B operand per lane)
__device__ __forceinline__ V16 load_b16x32(const unsigned char* base) {
  V16 B;
  const uint4* p = (const uint4*)base;
  uint4 b0 = p[0], b1 = p[2];
  B.u[0] = b0.x; B.u[1] = b0.y; B.u[2] = b0.z; B.u[3] = b0.w;
  B.u[4] = b1.x; B.u[5] = b1.y; B.u[6] = b1.z; B.u[7] = b1.w;
  return B;
}

struct AF { float4 a0, a1, a2, a3; };
__device__ __forceinline__ AF load_a_f32(const float* p0) {
  AF r;
  r.a0 = *(const float4*)(p0);
  r.a1 = *(const float4*)(p0 + 4);
  r.a2 = *(const float4*)(p0 + 16);
  r.a3 = *(const float4*)(p0 + 20);
  return r;
}
__device__ __forceinline__ V16 cvt_a(const AF& f) {
  V16 A;
  A.u[0] = f2bf2(f.a0.x, f.a0.y); A.u[1] = f2bf2(f.a0.z, f.a0.w);
  A.u[2] = f2bf2(f.a1.x, f.a1.y); A.u[3] = f2bf2(f.a1.z, f.a1.w);
  A.u[4] = f2bf2(f.a2.x, f.a2.y); A.u[5] = f2bf2(f.a2.z, f.a2.w);
  A.u[6] = f2bf2(f.a3.x, f.a3.y); A.u[7] = f2bf2(f.a3.z, f.a3.w);
  return A;
}

// explicit 2-deep software pipeline, order pinned by sched barriers
#define WMMA_PIPE_LOOP(N, ACC, LOADEXPR)                                        \
  {                                                                             \
    V16 B0 = (LOADEXPR(0));                                                     \
    V16 B1 = (LOADEXPR(1));                                                     \
    SCHED_FENCE();                                                              \
    _Pragma("unroll")                                                           \
    for (int it = 0; it < (N); it += 2) {                                       \
      ACC[it] = __builtin_amdgcn_wmma_f32_16x16x32_bf16(                        \
          false, A.v, false, B0.v, (short)0, ACC[it], false, false);            \
      if (it + 2 < (N)) B0 = (LOADEXPR(it + 2));                                \
      SCHED_FENCE();                                                            \
      ACC[it + 1] = __builtin_amdgcn_wmma_f32_16x16x32_bf16(                    \
          false, A.v, false, B1.v, (short)0, ACC[it + 1], false, false);        \
      if (it + 3 < (N)) B1 = (LOADEXPR(it + 3));                                \
      SCHED_FENCE();                                                            \
    }                                                                           \
  }

// =====================  pre-pass: x (f32) -> bf16 workspace  =====================
__global__ void __launch_bounds__(256)
cvt_f32_to_bf16(const float* __restrict__ x, unsigned short* __restrict__ xbf) {
  size_t i0 = ((size_t)blockIdx.x * 256 + threadIdx.x) * 8;
  float4 a = *(const float4*)(x + i0);
  float4 b = *(const float4*)(x + i0 + 4);
  uint4 o;
  o.x = f2bf2(a.x, a.y); o.y = f2bf2(a.z, a.w);
  o.z = f2bf2(b.x, b.y); o.w = f2bf2(b.z, b.w);
  *(uint4*)(xbf + i0) = o;
}

// =====================  main kernel, bf16 source + async staging  =====================
__global__ void __launch_bounds__(256)
dilated_attn_wmma_bf16(const unsigned short* __restrict__ xbf, float* __restrict__ out)
{
  extern __shared__ unsigned char smem[];
  unsigned short* pbuf  = (unsigned short*)smem;
  unsigned char*  kvbuf = smem + PBUF_BYTES;
  const unsigned  kv_lds = lds_offset_of(kvbuf);

  const int tid  = threadIdx.x;
  const int wave = tid >> 5;
  const int lane = tid & 31;
  const int nlo  = lane & 15;
  const int half = lane >> 4;

  const int bid = blockIdx.x;
  const int qt  = bid & 3;
  const int seg = (bid >> 2) & 3;
  const int off = (bid >> 4) & 3;
  const int b   = (bid >> 6) & 3;

  const unsigned short* xb = xbf + (size_t)b * 8192 * DHEAD;
  float*                ob = out + (size_t)b * 8192 * DHEAD;

  const int kvbase = off + 4 * (seg * 512);
  const int q0     = qt * 128 + wave * 16;

  // -------- Phase 1: S = Q K^T --------
  v8f acc[32];
#pragma unroll
  for (int i = 0; i < 32; ++i)
#pragma unroll
    for (int v = 0; v < 8; ++v) acc[i][v] = 0.0f;

  const unsigned short* qrow = xb + (size_t)(kvbase + 4 * (q0 + nlo)) * DHEAD;

  uint4 na0 = *(const uint4*)(qrow + 8 * half);        // A prefetch for kc=0
  uint4 na1 = *(const uint4*)(qrow + 16 + 8 * half);

  for (int kc = 0; kc < 32; ++kc) {
    __syncthreads();
#pragma unroll
    for (int i = 0; i < 16; ++i) {
      int flat = tid + i * 256;        // 4096 x 8-byte transfers
      int r    = flat >> 3;
      int kp8  = flat & 7;
      unsigned ldst = kv_lds + (unsigned)(r * 80 + kp8 * 8);
      const void* g = (const unsigned char*)xb
                    + ((size_t)(kvbase + 4 * r) * DHEAD + kc * 32) * 2 + kp8 * 8;
      async_ld_b64(ldst, g);
    }
    wait_async0();
    __syncthreads();

    V16 A;
    A.u[0] = na0.x; A.u[1] = na0.y; A.u[2] = na0.z; A.u[3] = na0.w;
    A.u[4] = na1.x; A.u[5] = na1.y; A.u[6] = na1.z; A.u[7] = na1.w;
    if (kc + 1 < 32) {
      na0 = *(const uint4*)(qrow + (kc + 1) * 32 + 8 * half);
      na1 = *(const uint4*)(qrow + (kc + 1) * 32 + 16 + 8 * half);
    }

#define LDB_P1(nt) load_b16x32(kvbuf + ((nt) * 16 + nlo) * 80 + half * 16)
    WMMA_PIPE_LOOP(32, acc, LDB_P1)
#undef LDB_P1
  }

  // -------- Softmax --------
  float rmax[8], rsum[8], rinv[8];
#pragma unroll
  for (int v = 0; v < 8; ++v) rmax[v] = -3.0e38f;
#pragma unroll
  for (int nt = 0; nt < 32; ++nt)
#pragma unroll
    for (int v = 0; v < 8; ++v) rmax[v] = fmaxf(rmax[v], acc[nt][v]);
#pragma unroll
  for (int m = 1; m < 16; m <<= 1)
#pragma unroll
    for (int v = 0; v < 8; ++v) rmax[v] = fmaxf(rmax[v], __shfl_xor(rmax[v], m, 32));
#pragma unroll
  for (int v = 0; v < 8; ++v) rsum[v] = 0.0f;

  unsigned short* pw = pbuf + wave * 16 * PB_STRIDE;
#pragma unroll
  for (int nt = 0; nt < 32; ++nt)
#pragma unroll
    for (int v = 0; v < 8; ++v) {
      float p = __expf(SCALE * (acc[nt][v] - rmax[v]));
      rsum[v] += p;
      pw[(v + 8 * half) * PB_STRIDE + nt * 16 + nlo] = f2bf(p);
    }
#pragma unroll
  for (int m = 1; m < 16; m <<= 1)
#pragma unroll
    for (int v = 0; v < 8; ++v) rsum[v] += __shfl_xor(rsum[v], m, 32);
#pragma unroll
  for (int v = 0; v < 8; ++v) rinv[v] = 0.25f / rsum[v];

  // -------- Phase 2: O = P V --------
  for (int dc = 0; dc < 8; ++dc) {
    v8f oacc[8];
#pragma unroll
    for (int j = 0; j < 8; ++j)
#pragma unroll
      for (int v = 0; v < 8; ++v) oacc[j][v] = 0.0f;

    for (int kt = 0; kt < 16; ++kt) {
      __syncthreads();
#pragma unroll
      for (int i = 0; i < 8; ++i) {
        int flat = tid + i * 256;      // 2048 uint (= 2 bf16) transfers
        int rr  = flat >> 6;
        int cp  = flat & 63;
        unsigned u = *(const unsigned*)(xb + (size_t)(kvbase + 4 * (kt * 32 + rr)) * DHEAD
                                           + dc * 128 + 2 * cp);
        *(unsigned short*)(kvbuf + (2 * cp) * 80 + rr * 2)     = (unsigned short)u;
        *(unsigned short*)(kvbuf + (2 * cp + 1) * 80 + rr * 2) = (unsigned short)(u >> 16);
      }
      __syncthreads();

      V16 A;
      {
        const unsigned short* pr = pw + nlo * PB_STRIDE;
        uint4 a0 = *(const uint4*)(pr + kt * 32 + 8 * half);
        uint4 a1 = *(const uint4*)(pr + kt * 32 + 16 + 8 * half);
        A.u[0] = a0.x; A.u[1] = a0.y; A.u[2] = a0.z; A.u[3] = a0.w;
        A.u[4] = a1.x; A.u[5] = a1.y; A.u[6] = a1.z; A.u[7] = a1.w;
      }

#define LDB_P2(j) load_b16x32(kvbuf + ((j) * 16 + nlo) * 80 + half * 16)
      WMMA_PIPE_LOOP(8, oacc, LDB_P2)
#undef LDB_P2
    }

#pragma unroll
    for (int j = 0; j < 8; ++j)
#pragma unroll
      for (int v = 0; v < 8; ++v) {
        size_t row = (size_t)(kvbase + 4 * (q0 + v + 8 * half));
        ob[row * DHEAD + dc * 128 + j * 16 + nlo] = oacc[j][v] * rinv[v];
      }
  }
}

// =====================  fallback: f32 source, in-loop conversion  =====================
__global__ void __launch_bounds__(256)
dilated_attn_wmma(const float* __restrict__ x, float* __restrict__ out)
{
  extern __shared__ unsigned char smem[];
  unsigned short* pbuf  = (unsigned short*)smem;
  unsigned char*  kvbuf = smem + PBUF_BYTES;

  const int tid  = threadIdx.x;
  const int wave = tid >> 5;
  const int lane = tid & 31;
  const int nlo  = lane & 15;
  const int half = lane >> 4;

  const int bid = blockIdx.x;
  const int qt  = bid & 3;
  const int seg = (bid >> 2) & 3;
  const int off = (bid >> 4) & 3;
  const int b   = (bid >> 6) & 3;

  const float* xb = x   + (size_t)b * 8192 * DHEAD;
  float*       ob = out + (size_t)b * 8192 * DHEAD;

  const int kvbase = off + 4 * (seg * 512);
  const int q0     = qt * 128 + wave * 16;

  v8f acc[32];
#pragma unroll
  for (int i = 0; i < 32; ++i)
#pragma unroll
    for (int v = 0; v < 8; ++v) acc[i][v] = 0.0f;

  const float* qrow = xb + (size_t)(kvbase + 4 * (q0 + nlo)) * DHEAD;
  AF afut = load_a_f32(qrow + 8 * half);

  for (int kc = 0; kc < 32; ++kc) {
    __syncthreads();
#pragma unroll 8
    for (int i = 0; i < 32; ++i) {
      int flat = tid + i * 256;
      int r  = flat >> 4;
      int kp = flat & 15;
      const float2 f = *(const float2*)(xb + (size_t)(kvbase + 4 * r) * DHEAD + kc * 32 + kp * 2);
      *((unsigned*)(kvbuf + r * 80) + kp) = f2bf2(f.x, f.y);
    }
    __syncthreads();

    V16 A = cvt_a(afut);
    if (kc + 1 < 32)
      afut = load_a_f32(qrow + (kc + 1) * 32 + 8 * half);

#define LDB_P1(nt) load_b16x32(kvbuf + ((nt) * 16 + nlo) * 80 + half * 16)
    WMMA_PIPE_LOOP(32, acc, LDB_P1)
#undef LDB_P1
  }

  float rmax[8], rsum[8], rinv[8];
#pragma unroll
  for (int v = 0; v < 8; ++v) rmax[v] = -3.0e38f;
#pragma unroll
  for (int nt = 0; nt < 32; ++nt)
#pragma unroll
    for (int v = 0; v < 8; ++v) rmax[v] = fmaxf(rmax[v], acc[nt][v]);
#pragma unroll
  for (int m = 1; m < 16; m <<= 1)
#pragma unroll
    for (int v = 0; v < 8; ++v) rmax[v] = fmaxf(rmax[v], __shfl_xor(rmax[v], m, 32));
#pragma unroll
  for (int v = 0; v < 8; ++v) rsum[v] = 0.0f;

  unsigned short* pw = pbuf + wave * 16 * PB_STRIDE;
#pragma unroll
  for (int nt = 0; nt < 32; ++nt)
#pragma unroll
    for (int v = 0; v < 8; ++v) {
      float p = __expf(SCALE * (acc[nt][v] - rmax[v]));
      rsum[v] += p;
      pw[(v + 8 * half) * PB_STRIDE + nt * 16 + nlo] = f2bf(p);
    }
#pragma unroll
  for (int m = 1; m < 16; m <<= 1)
#pragma unroll
    for (int v = 0; v < 8; ++v) rsum[v] += __shfl_xor(rsum[v], m, 32);
#pragma unroll
  for (int v = 0; v < 8; ++v) rinv[v] = 0.25f / rsum[v];

  for (int dc = 0; dc < 8; ++dc) {
    v8f oacc[8];
#pragma unroll
    for (int j = 0; j < 8; ++j)
#pragma unroll
      for (int v = 0; v < 8; ++v) oacc[j][v] = 0.0f;

    for (int kt = 0; kt < 16; ++kt) {
      __syncthreads();
#pragma unroll 8
      for (int i = 0; i < 16; ++i) {
        int flat = tid + i * 256;
        int rr = flat >> 7;
        int cc = flat & 127;
        float f = xb[(size_t)(kvbase + 4 * (kt * 32 + rr)) * DHEAD + dc * 128 + cc];
        *(unsigned short*)(kvbuf + cc * 80 + rr * 2) = f2bf(f);
      }
      __syncthreads();

      V16 A;
      {
        const unsigned short* pr = pw + nlo * PB_STRIDE;
        uint4 a0 = *(const uint4*)(pr + kt * 32 + 8 * half);
        uint4 a1 = *(const uint4*)(pr + kt * 32 + 16 + 8 * half);
        A.u[0] = a0.x; A.u[1] = a0.y; A.u[2] = a0.z; A.u[3] = a0.w;
        A.u[4] = a1.x; A.u[5] = a1.y; A.u[6] = a1.z; A.u[7] = a1.w;
      }

#define LDB_P2(j) load_b16x32(kvbuf + ((j) * 16 + nlo) * 80 + half * 16)
      WMMA_PIPE_LOOP(8, oacc, LDB_P2)
#undef LDB_P2
    }

#pragma unroll
    for (int j = 0; j < 8; ++j)
#pragma unroll
      for (int v = 0; v < 8; ++v) {
        size_t row = (size_t)(kvbase + 4 * (q0 + v + 8 * half));
        ob[row * DHEAD + dc * 128 + j * 16 + nlo] = oacc[j][v] * rinv[v];
      }
  }
}

extern "C" void kernel_launch(void* const* d_in, const int* in_sizes, int n_in,
                              void* d_out, int out_size, void* d_ws, size_t ws_size,
                              hipStream_t stream) {
  const float* x = (const float*)d_in[0];
  float* out = (float*)d_out;
  const size_t N = (size_t)4 * 8192 * 1024;      // 33,554,432 elements

  if (ws_size >= N * 2) {
    unsigned short* xbf = (unsigned short*)d_ws;
    cvt_f32_to_bf16<<<dim3((unsigned)(N / (256 * 8))), dim3(256), 0, stream>>>(x, xbf);
    dilated_attn_wmma_bf16<<<dim3(256), dim3(256), SMEM_BYTES, stream>>>(xbf, out);
  } else {
    dilated_attn_wmma<<<dim3(256), dim3(256), SMEM_BYTES, stream>>>(x, out);
  }
}
